// RGCN_v0_45243185496738
// MI455X (gfx1250) — matrix operand order repaired and checked by reading, not compiled, not run on hardware
//
#include <hip/hip_runtime.h>
#include <hip/hip_bf16.h>

#define NN 4096
#define TT 365
#define FF 16
#define HH 64
#define KSPLIT 4

typedef __attribute__((ext_vector_type(8)))  __bf16 v8bf;
typedef __attribute__((ext_vector_type(16))) __bf16 v16bf;
typedef __attribute__((ext_vector_type(8)))  float  v8f;

__device__ __forceinline__ unsigned short f32_to_bf16_bits(float f) {
  unsigned u = __builtin_bit_cast(unsigned, f);
  unsigned r = u + 0x7FFFu + ((u >> 16) & 1u);   // round-to-nearest-even
  return (unsigned short)(r >> 16);
}

__device__ __forceinline__ float sigmoidf_(float x) {
  return 1.0f / (1.0f + __expf(-x));
}

// A-fragment (16x32 bf16): this lane holds 16 consecutive K of its half
__device__ __forceinline__ v16bf load_afrag(const unsigned short* p) {
  v8bf lo = *reinterpret_cast<const v8bf*>(p);
  v8bf hi = *reinterpret_cast<const v8bf*>(p + 8);
  return __builtin_shufflevector(lo, hi, 0, 1, 2, 3, 4, 5, 6, 7,
                                 8, 9, 10, 11, 12, 13, 14, 15);
}
// B-fragment (32x16 bf16): this lane holds 16 consecutive K of its half
__device__ __forceinline__ v16bf load_bfrag(const unsigned short* p) {
  v8bf lo = *reinterpret_cast<const v8bf*>(p);
  v8bf hi = *reinterpret_cast<const v8bf*>(p + 8);
  return __builtin_shufflevector(lo, hi, 0, 1, 2, 3, 4, 5, 6, 7,
                                 8, 9, 10, 11, 12, 13, 14, 15);
}

__device__ __forceinline__ v8f wmma_bf16(v16bf a, v16bf b, v8f c) {
  return __builtin_amdgcn_wmma_f32_16x16x32_bf16(false, a, false, b,
                                                 (short)0, c, false, false);
}

// ---- A (f32) -> A_bf16, once per launch; bf16 A (32MB) stays L2-resident ----
__global__ void rgcn_convert_A(const float* __restrict__ A,
                               unsigned short* __restrict__ A16) {
  long i = (long)blockIdx.x * blockDim.x + threadIdx.x;
  const long total = (long)NN * NN;
  const long stride = (long)gridDim.x * blockDim.x;
  for (; i < total; i += stride) A16[i] = f32_to_bf16_bits(A[i]);
}

// ---- t=0 gates with h=c=0: graph-conv inputs collapse to the biases ----
__global__ void rgcn_gates0(const float* __restrict__ x,
                            const float* __restrict__ Wih,   // [16,256]
                            const float* __restrict__ bias,  // [256]
                            const float* __restrict__ bgh, const float* __restrict__ bgc,
                            float* __restrict__ hcur, float* __restrict__ ccur,
                            unsigned short* __restrict__ Bt) {  // [128,4096] bf16
  __shared__ float xs[FF], gs[4 * HH];
  const int n = blockIdx.x;
  const int j = threadIdx.x;   // 0..255
  if (j < FF) xs[j] = x[(long)n * TT * FF + j];   // t = 0
  __syncthreads();
  float acc = bias[j];
#pragma unroll
  for (int k = 0; k < FF; ++k) acc += xs[k] * Wih[k * 256 + j];
  gs[j] = acc;
  __syncthreads();
  if (j < HH) {
    float it = sigmoidf_(gs[j]);
    float gt = tanhf(gs[2 * HH + j]);
    ccur[(long)n * HH + j] = it * gt;   // f*0 + i*g
    hcur[(long)n * HH + j] = 0.0f;      // o*tanh(0)
    Bt[(long)j * NN + n] = f32_to_bf16_bits(bgh[j]);          // h=0 -> bias only
  } else if (j < 2 * HH) {
    Bt[(long)j * NN + n] = f32_to_bf16_bits(bgc[j - HH]);     // c=0 -> bias only
  }
}

// ---- K2: Gpart[kz] = A[:, K-chunk] @ B[K-chunk, 128]  (bf16 WMMA, f32 acc)
// 8 waves/block, each wave a 32x32 block (2x2 tiles): 8 b128 loads / 4 WMMAs.
// Block covers 64 rows x 128 cols; grid = (4096/64, KSPLIT) = 256 blocks.
__global__ void __launch_bounds__(256)
rgcn_graph_gemm(const unsigned short* __restrict__ A16,  // [4096,4096] bf16
                const unsigned short* __restrict__ Bt,   // [128,4096] bf16 (B^T)
                float* __restrict__ Gpart) {             // [KSPLIT][4096][128]
  const int lane = threadIdx.x & 31;
  const int wave = threadIdx.x >> 5;       // 0..7
  const int sel  = lane >> 4;              // half-wave select
  const int l15  = lane & 15;
  const int row0 = blockIdx.x * 64 + (wave >> 2) * 32;  // 32-row wave block
  const int col0 = (wave & 3) * 32;                     // 32-col wave block
  const int k0   = blockIdx.y * (NN / KSPLIT);          // K chunk base

  const unsigned short* a0p = A16 + (long)(row0 + l15) * NN + k0 + sel * 16;
  const unsigned short* a1p = a0p + 16L * NN;
  const unsigned short* b0p = Bt + (long)(col0 + l15) * NN + k0 + sel * 16;
  const unsigned short* b1p = b0p + 16L * NN;

  v8f acc00 = {}, acc01 = {}, acc10 = {}, acc11 = {};
#pragma unroll 2
  for (int k = 0; k < NN / KSPLIT; k += 32) {
    v16bf a0 = load_afrag(a0p + k);
    v16bf a1 = load_afrag(a1p + k);
    v16bf b0 = load_bfrag(b0p + k);
    v16bf b1 = load_bfrag(b1p + k);
    acc00 = wmma_bf16(a0, b0, acc00);
    acc01 = wmma_bf16(a0, b1, acc01);
    acc10 = wmma_bf16(a1, b0, acc10);
    acc11 = wmma_bf16(a1, b1, acc11);
  }

  float* G = Gpart + (long)blockIdx.y * NN * 128;
  // D layout: VGPR r -> M = r + 8*sel, N = lane&15
#pragma unroll
  for (int r = 0; r < 8; ++r) {
    const int rA = row0 + sel * 8 + r;
    const int rB = rA + 16;
    G[(long)rA * 128 + col0 + l15]      = acc00[r];
    G[(long)rA * 128 + col0 + 16 + l15] = acc01[r];
    G[(long)rB * 128 + col0 + l15]      = acc10[r];
    G[(long)rB * 128 + col0 + 16 + l15] = acc11[r];
  }
}

// ---- K3: combine(t) fused with gates(t+1); hidden state never leaves LDS ----
// one block (256 threads) per node
__global__ void rgcn_update(const float* __restrict__ Gpart,  // [KSPLIT][4096][128]
                            const float* __restrict__ Whc, const float* __restrict__ Whp,
                            const float* __restrict__ bh,
                            const float* __restrict__ Wcc, const float* __restrict__ Wcp,
                            const float* __restrict__ bc,
                            const float* __restrict__ dw, const float* __restrict__ db,
                            const float* __restrict__ x, int t, int do_next,
                            const float* __restrict__ Wih,   // [16,256]
                            const float* __restrict__ Whh,   // [64,256]
                            const float* __restrict__ bias,  // [256]
                            const float* __restrict__ Wgh, const float* __restrict__ bgh,
                            const float* __restrict__ Wgc, const float* __restrict__ bgc,
                            float* __restrict__ hcur, float* __restrict__ ccur,
                            unsigned short* __restrict__ Bt,
                            float* __restrict__ out) {
  __shared__ float xs[FF], hs[HH], cs[HH], gs[4 * HH];
  __shared__ float hc_s[HH], cc_s[HH], hg_s[HH], cg_s[HH];
  const int n = blockIdx.x;
  const int j = threadIdx.x;  // 0..255
  const long gbase = (long)n * 128;
  const long gstep = (long)NN * 128;

  // stage combine inputs (j<128) and next-step x (j in 160..175)
  if (j < HH) {
    hc_s[j] = hcur[(long)n * HH + j];
    float g = Gpart[gbase + j];
#pragma unroll
    for (int p = 1; p < KSPLIT; ++p) g += Gpart[gbase + p * gstep + j];
    hg_s[j] = tanhf(g);
  } else if (j < 2 * HH) {
    const int jj = j - HH;
    cc_s[jj] = ccur[(long)n * HH + jj];
    float g = Gpart[gbase + HH + jj];
#pragma unroll
    for (int p = 1; p < KSPLIT; ++p) g += Gpart[gbase + p * gstep + HH + jj];
    cg_s[jj] = tanhf(g);
  } else if (do_next && j >= 160 && j < 160 + FF) {
    xs[j - 160] = x[((long)n * TT + (t + 1)) * FF + (j - 160)];
  }
  __syncthreads();

  // combine: h_new / c_new (stay in LDS)
  if (j < HH) {
    float a = bh[j];
#pragma unroll 8
    for (int k = 0; k < HH; ++k)
      a += hc_s[k] * Whc[k * HH + j] + hg_s[k] * Whp[k * HH + j];
    hs[j] = sigmoidf_(a);
  } else if (j < 2 * HH) {
    const int jj = j - HH;
    float a = bc[jj];
#pragma unroll 8
    for (int k = 0; k < HH; ++k)
      a += cc_s[k] * Wcc[k * HH + jj] + cg_s[k] * Wcp[k * HH + jj];
    cs[jj] = sigmoidf_(a);
  }
  __syncthreads();

  // per-timestep output: out[n,t] = h_new . dense_w + dense_b
  if (j == 0) {
    float o = db[0];
#pragma unroll 8
    for (int k = 0; k < HH; ++k) o += hs[k] * dw[k];
    out[(long)n * TT + t] = o;
  }

  if (!do_next) return;   // block-uniform

  // gates for t+1 from h_new/c_new in LDS
  float acc = bias[j];
#pragma unroll
  for (int k = 0; k < FF; ++k) acc += xs[k] * Wih[k * 256 + j];
#pragma unroll 8
  for (int k = 0; k < HH; ++k) acc += hs[k] * Whh[k * 256 + j];
  gs[j] = acc;
  __syncthreads();

  if (j < HH) {
    float it = sigmoidf_(gs[j]);
    float ft = sigmoidf_(gs[HH + j]);
    float gt = tanhf(gs[2 * HH + j]);
    float ot = sigmoidf_(gs[3 * HH + j]);
    ccur[(long)n * HH + j] = ft * cs[j] + it * gt;
    hcur[(long)n * HH + j] = ot * tanhf(cs[j]);     // tanh of c_new(t) = c_prev of t+1
    float hg = bgh[j];
#pragma unroll 8
    for (int k = 0; k < HH; ++k) hg += hs[k] * Wgh[k * HH + j];
    Bt[(long)j * NN + n] = f32_to_bf16_bits(hg);
  } else if (j < 2 * HH) {
    const int jj = j - HH;
    float cg = bgc[jj];
#pragma unroll 8
    for (int k = 0; k < HH; ++k) cg += cs[k] * Wgc[k * HH + jj];
    Bt[(long)(HH + jj) * NN + n] = f32_to_bf16_bits(cg);
  }
}

extern "C" void kernel_launch(void* const* d_in, const int* in_sizes, int n_in,
                              void* d_out, int out_size, void* d_ws, size_t ws_size,
                              hipStream_t stream) {
  const float* x    = (const float*)d_in[0];
  const float* A    = (const float*)d_in[1];
  const float* Wih  = (const float*)d_in[2];
  const float* Whh  = (const float*)d_in[3];
  const float* bias = (const float*)d_in[4];
  const float* Wgh  = (const float*)d_in[5];
  const float* bgh  = (const float*)d_in[6];
  const float* Wgc  = (const float*)d_in[7];
  const float* bgc  = (const float*)d_in[8];
  const float* Whc  = (const float*)d_in[9];
  const float* Whp  = (const float*)d_in[10];
  const float* bh   = (const float*)d_in[11];
  const float* Wcc  = (const float*)d_in[12];
  const float* Wcp  = (const float*)d_in[13];
  const float* bc   = (const float*)d_in[14];
  const float* dw   = (const float*)d_in[15];
  const float* db   = (const float*)d_in[16];
  float* out = (float*)d_out;

  char* ws = (char*)d_ws;
  const size_t MB = 1u << 20;
  unsigned short* A16 = (unsigned short*)(ws);            // 32 MB  bf16 adjacency
  unsigned short* Bt  = (unsigned short*)(ws + 32 * MB);  //  1 MB  [128,4096] bf16
  float* hcur  = (float*)(ws + 33 * MB);                  //  1 MB
  float* ccur  = (float*)(ws + 34 * MB);                  //  1 MB
  float* Gpart = (float*)(ws + 35 * MB);                  //  8 MB [KSPLIT][4096][128]
  (void)in_sizes; (void)n_in; (void)out_size; (void)ws_size;

  rgcn_convert_A<<<4096, 256, 0, stream>>>(A, A16);
  rgcn_gates0<<<NN, 256, 0, stream>>>(x, Wih, bias, bgh, bgc, hcur, ccur, Bt);

  const dim3 gemm_grid(NN / 64, KSPLIT);   // 256 blocks
  for (int t = 0; t < TT; ++t) {
    rgcn_graph_gemm<<<gemm_grid, 256, 0, stream>>>(A16, Bt, Gpart);
    rgcn_update<<<NN, 256, 0, stream>>>(Gpart, Whc, Whp, bh, Wcc, Wcp, bc,
                                        dw, db, x, t, (t + 1 < TT) ? 1 : 0,
                                        Wih, Whh, bias, Wgh, bgh, Wgc, bgc,
                                        hcur, ccur, Bt, out);
  }
}
